// GQA_51565377356263
// MI455X (gfx1250) — compile-verified, matrix operand order checked by reading, not run on hardware
//
#include <hip/hip_runtime.h>
#include <hip/hip_bf16.h>
#include <cstdint>

// ---------------------------------------------------------------------------
// GQA forward for MI455X (gfx1250, wave32, WMMA).
//   fp32 -> bf16 demotion; Q/K/V GEMMs + flash attention + O-proj,
//   all matmuls via v_wmma_f32_16x16x32_bf16 (f32 accumulate).
//   - fragments assembled with aligned 32-bit / b128 LDS reads
//   - softmax reductions via DPP16 butterflies (no LDS round-trips)
//   - raw v_max_num_f32 via inline asm (avoids canonicalize self-max)
//   - 1/sqrt(d) folded into the Q-projection epilogue
// ---------------------------------------------------------------------------

#define D_MODEL   4096
#define SEQ       2048
#define N_HEADS   64
#define D_HEAD    64
#define D_GROUP   512

typedef __attribute__((ext_vector_type(16))) __bf16 v16bf;
typedef __attribute__((ext_vector_type(8)))  float  v8f;

union Frag16 {
    v16bf v;
    unsigned int w[8];      // 32-bit fill: w[j] = elements (2j, 2j+1)
};

union U4x8 {
    uint4 v;
    unsigned short s[8];
};

// float -> bf16, round-to-nearest-even
__device__ __forceinline__ unsigned short f2bf(float f) {
    unsigned int u = __float_as_uint(f);
    u += 0x7FFFu + ((u >> 16) & 1u);
    return (unsigned short)(u >> 16);
}

// CDNA5 16-bit A/B layout (ISA 7.12.2): VGPR j of lane L (hi = L>=16) holds
// K indices k, k+1 with k = 16*(j>=4) + 8*hi + 2*(j&3).  Always even -> the
// pair is one aligned 32-bit read when K is the contiguous dimension.
__device__ __forceinline__ int frag_k2(int hi, int j) {
    return ((j & 4) << 2) + (hi << 3) + ((j & 3) << 1);
}

// Raw IEEE maximumNumber without the compiler's canonicalize self-max.
__device__ __forceinline__ float fmax_raw(float a, float b) {
    float d;
    asm("v_max_num_f32 %0, %1, %2" : "=v"(d) : "v"(a), "v"(b));
    return d;
}

// ---- DPP16 butterfly reduction within each 16-lane row (wave32 halves) ----
// xor1 = quad_perm(1,0,3,2)=0xB1, xor2 = quad_perm(2,3,0,1)=0x4E,
// xor7 = row_half_mirror=0x141, xor15 = row_mirror=0x140.
template <int CTRL>
__device__ __forceinline__ float dpp_movf(float x) {
    return __builtin_bit_cast(float,
        __builtin_amdgcn_update_dpp(0, __builtin_bit_cast(int, x),
                                    CTRL, 0xF, 0xF, true));
}
__device__ __forceinline__ float row_max16(float x) {
    x = fmax_raw(x, dpp_movf<0xB1>(x));
    x = fmax_raw(x, dpp_movf<0x4E>(x));
    x = fmax_raw(x, dpp_movf<0x141>(x));
    x = fmax_raw(x, dpp_movf<0x140>(x));
    return x;
}
__device__ __forceinline__ float row_sum16(float x) {
    x += dpp_movf<0xB1>(x);
    x += dpp_movf<0x4E>(x);
    x += dpp_movf<0x141>(x);
    x += dpp_movf<0x140>(x);
    return x;
}

// ---------------------------------------------------------------------------
// fp32 -> bf16 conversion (grid-stride)
// ---------------------------------------------------------------------------
__global__ void cvt_f32_bf16(const float* __restrict__ in,
                             unsigned short* __restrict__ out, long long n) {
    long long i = (long long)blockIdx.x * blockDim.x + threadIdx.x;
    long long stride = (long long)gridDim.x * blockDim.x;
    for (; i < n; i += stride) out[i] = f2bf(in[i]);
}

// ---------------------------------------------------------------------------
// bf16 GEMM: C[M,N] = oscale * (A[M,K] @ B[K,N] + bias).  f32 accumulate.
// Block = 256 thr (8 waves) -> 128x64 tile; wave grid 4(M) x 2(N), each wave
// a 32x32 sub-tile (4 accumulators, 2x2 fragment reuse, 4 WMMAs / K-step).
// K staged in 32-wide LDS tiles: A row-major, B transposed ([n][k]) so every
// fragment VGPR is one aligned 32-bit LDS read.
// ---------------------------------------------------------------------------
template <bool OUT_BF16>
__global__ __launch_bounds__(256)
void gemm_bf16_wmma(const unsigned short* __restrict__ A,
                    const unsigned short* __restrict__ B,
                    const float* __restrict__ bias,
                    void* __restrict__ Cout,
                    int M, int N, int K, float oscale) {
    __shared__ __align__(16) unsigned short As[128 * 32];   // [m][k]
    __shared__ __align__(16) unsigned short Bt[64 * 32];    // [n][k] (transposed)

    const int t    = threadIdx.x;
    const int wave = t >> 5;
    const int lane = t & 31;
    const int hi   = lane >> 4;
    const int ln   = lane & 15;

    const int waveM = (wave & 3) * 32;
    const int waveN = (wave >> 2) * 32;
    const int gm = blockIdx.y * 128;
    const int gn = blockIdx.x * 64;

    v8f c00 = {0.f,0.f,0.f,0.f,0.f,0.f,0.f,0.f};
    v8f c01 = c00, c10 = c00, c11 = c00;

    const int arow = t >> 2, acol = (t & 3) * 8;   // A: 2x (64 rows x 32 cols)
    const int brow = t >> 3, bcol = (t & 7) * 8;   // B: 32 rows(k) x 64 cols(n)

    for (int k0 = 0; k0 < K; k0 += 32) {
        __syncthreads();
        *(uint4*)(As + arow * 32 + acol) =
            *(const uint4*)(A + (size_t)(gm + arow) * K + k0 + acol);
        *(uint4*)(As + (64 + arow) * 32 + acol) =
            *(const uint4*)(A + (size_t)(gm + 64 + arow) * K + k0 + acol);
        {
            U4x8 u;
            u.v = *(const uint4*)(B + (size_t)(k0 + brow) * N + gn + bcol);
#pragma unroll
            for (int j = 0; j < 8; ++j)
                Bt[(bcol + j) * 32 + brow] = u.s[j];
        }
        __syncthreads();

        if (k0 + 32 < K) {  // prefetch next K-tile -> global_prefetch_b8
            __builtin_prefetch(A + (size_t)(gm + arow) * K + k0 + 32 + acol, 0, 1);
            __builtin_prefetch(A + (size_t)(gm + 64 + arow) * K + k0 + 32 + acol, 0, 1);
            __builtin_prefetch(B + (size_t)(k0 + 32 + brow) * N + gn + bcol, 0, 1);
        }

        Frag16 a0, a1, b0, b1;
#pragma unroll
        for (int j = 0; j < 8; ++j) {
            int k = frag_k2(hi, j);
            a0.w[j] = *(const unsigned int*)(As + (waveM + ln) * 32 + k);
            a1.w[j] = *(const unsigned int*)(As + (waveM + 16 + ln) * 32 + k);
            b0.w[j] = *(const unsigned int*)(Bt + (waveN + ln) * 32 + k);
            b1.w[j] = *(const unsigned int*)(Bt + (waveN + 16 + ln) * 32 + k);
        }
        c00 = __builtin_amdgcn_wmma_f32_16x16x32_bf16(false, a0.v, false, b0.v,
                                                      (short)0, c00, false, false);
        c01 = __builtin_amdgcn_wmma_f32_16x16x32_bf16(false, a0.v, false, b1.v,
                                                      (short)0, c01, false, false);
        c10 = __builtin_amdgcn_wmma_f32_16x16x32_bf16(false, a1.v, false, b0.v,
                                                      (short)0, c10, false, false);
        c11 = __builtin_amdgcn_wmma_f32_16x16x32_bf16(false, a1.v, false, b1.v,
                                                      (short)0, c11, false, false);
    }

    // epilogue: C/D layout -> lane L, VGPR r => row r+8*hi, col ln
    v8f* acc[4] = {&c00, &c01, &c10, &c11};
#pragma unroll
    for (int q = 0; q < 4; ++q) {
        int mi = q >> 1, ni = q & 1;
#pragma unroll
        for (int r = 0; r < 8; ++r) {
            int row = gm + waveM + mi * 16 + r + hi * 8;
            int col = gn + waveN + ni * 16 + ln;
            float v = ((*acc[q])[r] + bias[col]) * oscale;
            if (OUT_BF16)
                ((unsigned short*)Cout)[(size_t)row * N + col] = f2bf(v);
            else
                ((float*)Cout)[(size_t)row * N + col] = v;
        }
    }
}

// ---------------------------------------------------------------------------
// Flash attention (non-causal), bf16 in / bf16 out, f32 accumulate.
// Block = 8 waves (one head, 128 query rows); wave owns 16 query rows.
// Q is pre-scaled by 1/sqrt(d) in its projection.  Per 32-key block the K
// tile ([key][dim]) and V tile (transposed [dim][key]) are staged
// cooperatively in LDS, then: 4 WMMAs for scores, streaming softmax with
// DPP16 butterfly reductions, P bounced through LDS (C->A relayout),
// 4 WMMAs for P@V.
// ---------------------------------------------------------------------------
__global__ __launch_bounds__(256)
void flash_attn_wmma(const unsigned short* __restrict__ Qb,   // [S, D_MODEL]
                     const unsigned short* __restrict__ Kb,   // [S, D_GROUP]
                     const unsigned short* __restrict__ Vb,   // [S, D_GROUP]
                     unsigned short* __restrict__ Ob) {       // [S, D_MODEL]
    __shared__ __align__(16) unsigned short Ks[32 * 64];      // [key][dim]
    __shared__ __align__(16) unsigned short Vt[64 * 32];      // [dim][key]
    __shared__ __align__(16) unsigned short Ps[8][16 * 32];   // per-wave P tile

    const int h    = blockIdx.y;
    const int g    = h >> 3;
    const int t    = threadIdx.x;
    const int wave = t >> 5;
    const int lane = t & 31;
    const int hi   = lane >> 4;
    const int ln   = lane & 15;
    const int q0   = blockIdx.x * 128 + wave * 16;

    const int srow = t >> 3, scol = (t & 7) * 8;  // staging: 32 keys x 64 dims

    // Q fragments: 16x64 -> two 16x32 A-frags (32-bit global reads)
    Frag16 qf[2];
    {
        const unsigned short* qr = Qb + (size_t)(q0 + ln) * D_MODEL + h * D_HEAD;
#pragma unroll
        for (int c = 0; c < 2; ++c)
#pragma unroll
            for (int j = 0; j < 8; ++j)
                qf[c].w[j] = *(const unsigned int*)(qr + c * 32 + frag_k2(hi, j));
    }

    float mrow[8], lrow[8];
    v8f o[4];
#pragma unroll
    for (int r = 0; r < 8; ++r) { mrow[r] = -1e30f; lrow[r] = 0.f; }
#pragma unroll
    for (int j = 0; j < 4; ++j) o[j] = v8f{0.f,0.f,0.f,0.f,0.f,0.f,0.f,0.f};

    unsigned short* pw = &Ps[wave][0];

    for (int t0 = 0; t0 < SEQ; t0 += 32) {
        // ---- cooperative K/V staging ----
        __syncthreads();
        *(uint4*)(Ks + srow * 64 + scol) =
            *(const uint4*)(Kb + (size_t)(t0 + srow) * D_GROUP + g * D_HEAD + scol);
        {
            U4x8 u;
            u.v = *(const uint4*)(Vb + (size_t)(t0 + srow) * D_GROUP + g * D_HEAD + scol);
#pragma unroll
            for (int j = 0; j < 8; ++j)
                Vt[(scol + j) * 32 + srow] = u.s[j];
        }
        __syncthreads();

        if (t0 + 32 < SEQ) {
            __builtin_prefetch(Kb + (size_t)(t0 + 32 + srow) * D_GROUP + g * D_HEAD + scol, 0, 1);
            __builtin_prefetch(Vb + (size_t)(t0 + 32 + srow) * D_GROUP + g * D_HEAD + scol, 0, 1);
        }

        // ---- scores: S = Q(16x64) @ K^T(64x32), two 16x16 tiles ----
        v8f slo = v8f{0.f,0.f,0.f,0.f,0.f,0.f,0.f,0.f};
        v8f shi = slo;
#pragma unroll
        for (int kb = 0; kb < 2; ++kb) {
            v8f acc = kb ? shi : slo;
#pragma unroll
            for (int c = 0; c < 2; ++c) {
                Frag16 b;   // B(kdim, n): lane -> key n, 32-bit reads along kdim
#pragma unroll
                for (int j = 0; j < 8; ++j)
                    b.w[j] = *(const unsigned int*)(
                        Ks + (kb * 16 + ln) * 64 + c * 32 + frag_k2(hi, j));
                acc = __builtin_amdgcn_wmma_f32_16x16x32_bf16(
                          false, qf[c].v, false, b.v, (short)0, acc, false, false);
            }
            if (kb) shi = acc; else slo = acc;
        }

        // ---- streaming softmax update (DPP16 butterflies, no LDS) ----
        float alpha[8];
#pragma unroll
        for (int r = 0; r < 8; ++r) {
            float a0 = slo[r];
            float a1 = shi[r];
            float bm = row_max16(fmax_raw(a0, a1));
            float nm = fmax_raw(mrow[r], bm);
            alpha[r] = __expf(mrow[r] - nm);
            mrow[r]  = nm;
            float p0 = __expf(a0 - nm);
            float p1 = __expf(a1 - nm);
            float rs = row_sum16(p0 + p1);
            lrow[r] = lrow[r] * alpha[r] + rs;
            int row = r + hi * 8;
            pw[row * 32 + ln]      = f2bf(p0);
            pw[row * 32 + 16 + ln] = f2bf(p1);
        }
#pragma unroll
        for (int j = 0; j < 4; ++j)
#pragma unroll
            for (int r = 0; r < 8; ++r) o[j][r] *= alpha[r];

        // ---- O += P(16x32) @ V(32x64); wave-private LDS, DS is in-order ----
        Frag16 pa;
#pragma unroll
        for (int j = 0; j < 8; ++j)
            pa.w[j] = *(const unsigned int*)(pw + ln * 32 + frag_k2(hi, j));
#pragma unroll
        for (int jt = 0; jt < 4; ++jt) {
            Frag16 vb;   // B(key, dim): lane -> dim jt*16+ln, 32-bit reads along key
#pragma unroll
            for (int j = 0; j < 8; ++j)
                vb.w[j] = *(const unsigned int*)(
                    Vt + (jt * 16 + ln) * 32 + frag_k2(hi, j));
            o[jt] = __builtin_amdgcn_wmma_f32_16x16x32_bf16(
                        false, pa.v, false, vb.v, (short)0, o[jt], false, false);
        }
    }

    // ---- epilogue: normalize, emit bf16 ----
#pragma unroll
    for (int jt = 0; jt < 4; ++jt)
#pragma unroll
        for (int r = 0; r < 8; ++r) {
            int row = q0 + r + hi * 8;
            float val = o[jt][r] / lrow[r];
            Ob[(size_t)row * D_MODEL + h * D_HEAD + jt * 16 + ln] = f2bf(val);
        }
}

// ---------------------------------------------------------------------------
// Host orchestration
// ---------------------------------------------------------------------------
static inline size_t align256(size_t x) { return (x + 255) & ~(size_t)255; }

extern "C" void kernel_launch(void* const* d_in, const int* in_sizes, int n_in,
                              void* d_out, int out_size, void* d_ws, size_t ws_size,
                              hipStream_t stream) {
    (void)in_sizes; (void)n_in; (void)out_size; (void)ws_size;

    const float* x  = (const float*)d_in[0];
    const float* wq = (const float*)d_in[1];
    const float* bq = (const float*)d_in[2];
    const float* wk = (const float*)d_in[3];
    const float* bk = (const float*)d_in[4];
    const float* wv = (const float*)d_in[5];
    const float* bv = (const float*)d_in[6];
    const float* wo = (const float*)d_in[7];
    const float* bo = (const float*)d_in[8];

    char* ws = (char*)d_ws;
    size_t off = 0;
    auto carve = [&](size_t bytes) { char* p = ws + off; off += align256(bytes); return p; };

    const size_t nX  = (size_t)SEQ * D_MODEL;
    const size_t nWq = (size_t)D_MODEL * D_MODEL;
    const size_t nWk = (size_t)D_MODEL * D_GROUP;

    unsigned short* xb  = (unsigned short*)carve(nX  * 2);
    unsigned short* wqb = (unsigned short*)carve(nWq * 2);
    unsigned short* wkb = (unsigned short*)carve(nWk * 2);
    unsigned short* wvb = (unsigned short*)carve(nWk * 2);
    unsigned short* wob = (unsigned short*)carve(nWq * 2);
    unsigned short* Qb  = (unsigned short*)carve(nX  * 2);
    unsigned short* Kb  = (unsigned short*)carve((size_t)SEQ * D_GROUP * 2);
    unsigned short* Vb  = (unsigned short*)carve((size_t)SEQ * D_GROUP * 2);
    unsigned short* Ab  = xb;  // x dead after QKV projections; reuse for attn out

    // --- fp32 -> bf16 conversions ---
    const int CT = 256;
    auto cgrid = [](long long n) { long long b = (n + 255) / 256; return (unsigned)(b > 16384 ? 16384 : b); };
    cvt_f32_bf16<<<cgrid(nX),  CT, 0, stream>>>(x,  xb,  (long long)nX);
    cvt_f32_bf16<<<cgrid(nWq), CT, 0, stream>>>(wq, wqb, (long long)nWq);
    cvt_f32_bf16<<<cgrid(nWk), CT, 0, stream>>>(wk, wkb, (long long)nWk);
    cvt_f32_bf16<<<cgrid(nWk), CT, 0, stream>>>(wv, wvb, (long long)nWk);
    cvt_f32_bf16<<<cgrid(nWq), CT, 0, stream>>>(wo, wob, (long long)nWq);

    // --- projections (bf16 out); 1/sqrt(d_head)=0.125 folded into Q ---
    gemm_bf16_wmma<true><<<dim3(D_MODEL / 64, SEQ / 128), 256, 0, stream>>>(
        xb, wqb, bq, Qb, SEQ, D_MODEL, D_MODEL, 0.125f);
    gemm_bf16_wmma<true><<<dim3(D_GROUP / 64, SEQ / 128), 256, 0, stream>>>(
        xb, wkb, bk, Kb, SEQ, D_GROUP, D_MODEL, 1.0f);
    gemm_bf16_wmma<true><<<dim3(D_GROUP / 64, SEQ / 128), 256, 0, stream>>>(
        xb, wvb, bv, Vb, SEQ, D_GROUP, D_MODEL, 1.0f);

    // --- flash attention ---
    flash_attn_wmma<<<dim3(SEQ / 128, N_HEADS), 256, 0, stream>>>(Qb, Kb, Vb, Ab);

    // --- output projection (fp32 out) ---
    gemm_bf16_wmma<false><<<dim3(D_MODEL / 64, SEQ / 128), 256, 0, stream>>>(
        Ab, wob, bo, d_out, SEQ, D_MODEL, D_MODEL, 1.0f);
}
